// ImageEncoderBlock_13254269075864
// MI455X (gfx1250) — matrix operand order repaired and checked by reading, not compiled, not run on hardware
//
#include <hip/hip_runtime.h>
#include <hip/hip_bf16.h>
#include <math.h>

// ---------------- types ----------------
typedef __bf16 bf16_t;
typedef __attribute__((ext_vector_type(16))) __bf16 v16bf;
typedef __attribute__((ext_vector_type(8)))  float  v8f;
typedef unsigned int u32x4 __attribute__((ext_vector_type(4)));
typedef int          i32x8 __attribute__((ext_vector_type(8)));
typedef int          i32x4 __attribute__((ext_vector_type(4)));

#if __has_builtin(__builtin_amdgcn_tensor_load_to_lds) && \
    __has_builtin(__builtin_amdgcn_s_wait_tensorcnt)
#define USE_TDM 1
#endif

// ---------------- problem constants ----------------
constexpr int Cc    = 768;
constexpr int C3    = 2304;
constexpr int HID   = 3072;
constexpr int HEADS = 12;
constexpr int HD    = 64;
constexpr int WS    = 14;
constexpr int LWIN  = 196;
constexpr int NWIN  = 100;
constexpr int NWH   = NWIN * HEADS;   // 1200
constexpr int MQKV  = NWIN * LWIN;    // 19600
constexpr int MTOK  = 4 * 64 * 64;    // 16384
constexpr int LP    = 224;            // padded window length (14x16, 7x32)

// LDS leading dims: lane stride mod 64 banks == 4k -> conflict-free b128
constexpr int LDT   = 40;             // GEMM tiles (80B rows: 64B payload + 16B pad)
constexpr int LDQ   = 72;             // sQ/sK
constexpr int LDV   = 232;            // sVt
constexpr int LDS_S = 228;            // sS f32

// =====================================================================
// WMMA fragment loaders (CDNA5 ISA 7.12.2, wave32) — contiguous b128
// =====================================================================
__device__ __forceinline__ v16bf ldA(const bf16_t* __restrict__ t, int ld) {
  int lane = threadIdx.x & 31;
  int r  = lane & 15;
  int kb = (lane >> 4) << 3;
  const bf16_t* row = t + r * ld;
  v16bf a;
#pragma unroll
  for (int i = 0; i < 8; ++i) a[i]     = row[kb + i];
#pragma unroll
  for (int i = 0; i < 8; ++i) a[8 + i] = row[kb + 16 + i];
  return a;
}
__device__ __forceinline__ v16bf ldA_f32(const float* __restrict__ t, int ld) {
  int lane = threadIdx.x & 31;
  int r  = lane & 15;
  int kb = (lane >> 4) << 3;
  const float* row = t + r * ld;
  v16bf a;
#pragma unroll
  for (int i = 0; i < 8; ++i) a[i]     = (bf16_t)row[kb + i];
#pragma unroll
  for (int i = 0; i < 8; ++i) a[8 + i] = (bf16_t)row[kb + 16 + i];
  return a;
}
__device__ __forceinline__ v16bf ldB_T(const bf16_t* __restrict__ t, int ld) {
  int lane = threadIdx.x & 31;
  int n  = lane & 15;
  int kb = (lane >> 4) << 3;
  const bf16_t* row = t + n * ld + kb;
  v16bf b;
#pragma unroll
  for (int i = 0; i < 8; ++i) b[i] = row[i];
#pragma unroll
  for (int i = 0; i < 8; ++i) b[8 + i] = row[16 + i];
  return b;
}
__device__ __forceinline__ v8f wmma_bf16(v16bf a, v16bf b, v8f c) {
  return __builtin_amdgcn_wmma_f32_16x16x32_bf16(false, a, false, b, (short)0, c,
                                                 false, false);
}

#ifdef USE_TDM
// =====================================================================
// Tensor Data Mover: 2D bf16 tile (tile_d0 x tile_d1) global -> LDS,
// LDS rows padded (pad_interval/pad_amount per D# spec 8.4).
// lds_byte_off is the offset inside the workgroup LDS segment.
// amdgpu-toolchain (clang-23) 6-arg builtin form.
// =====================================================================
__device__ __forceinline__ void tdm_load_2d(
    unsigned int lds_byte_off, const bf16_t* gsrc,
    unsigned int tensor_d0, unsigned int tensor_d1,
    unsigned long long stride_elems,
    unsigned int tile_d0, unsigned int tile_d1,
    unsigned int pad_interval, unsigned int pad_amount) {
  unsigned long long ga = (unsigned long long)(uintptr_t)gsrc;
  u32x4 g0;
  g0[0] = 1u;                                          // count=1, user mode
  g0[1] = lds_byte_off;                                // lds_addr (bytes)
  g0[2] = (unsigned int)ga;                            // global_addr[31:0]
  g0[3] = (unsigned int)((ga >> 32) & 0x01ffffffu) | (2u << 30);  // [56:32]|type=2
  i32x8 g1;
  g1[0] = (int)((1u << 16)                             // data_size = 2 bytes
              | (1u << 20)                             // pad_enable
              | (pad_interval << 22) | (pad_amount << 25));
  g1[1] = (int)((tensor_d0 & 0xffffu) << 16);          // tensor_dim0[15:0]
  g1[2] = (int)((tensor_d0 >> 16) | ((tensor_d1 & 0xffffu) << 16));
  g1[3] = (int)((tensor_d1 >> 16) | (tile_d0 << 16));  // | tile_dim0
  g1[4] = (int)(tile_d1 & 0xffffu);                    // tile_dim1, tile_dim2=0
  g1[5] = (int)(unsigned int)(stride_elems & 0xffffffffu);   // dim0_stride lo
  g1[6] = (int)(unsigned int)((stride_elems >> 32) & 0xffffu);
  g1[7] = 0;
  i32x4 z  = {0, 0, 0, 0};
  i32x8 z8 = {0, 0, 0, 0, 0, 0, 0, 0};
  __builtin_amdgcn_tensor_load_to_lds(g0, g1, z, z, z8, 0);
}
#endif

// =====================================================================
// fused transpose + fp32->bf16 : out(N x K) = in(K x N)^T   (one-time)
// =====================================================================
__global__ __launch_bounds__(256) void transpose_cvt_kernel(
    const float* __restrict__ in, bf16_t* __restrict__ out, int K, int N) {
  __shared__ float tile[32][33];
  int bx = blockIdx.x, by = blockIdx.y;
  int tx = threadIdx.x & 31, ty = threadIdx.x >> 5;
#pragma unroll
  for (int i = 0; i < 32; i += 8)
    tile[ty + i][tx] = in[(size_t)(by * 32 + ty + i) * N + bx * 32 + tx];
  __syncthreads();
#pragma unroll
  for (int i = 0; i < 32; i += 8)
    out[(size_t)(bx * 32 + ty + i) * K + by * 32 + tx] = (bf16_t)tile[tx][ty + i];
}

// =====================================================================
// LayerNorm1 + window partition  (block per windowed token row)
// =====================================================================
__global__ __launch_bounds__(256) void ln1_window_kernel(
    const float* __restrict__ x, const float* __restrict__ sc,
    const float* __restrict__ bi, bf16_t* __restrict__ out) {
  int r = blockIdx.x;
  int wi = r / LWIN, t = r - wi * LWIN;
  int b  = wi / 25, wrem = wi % 25;
  int y  = (wrem / 5) * WS + t / WS;
  int xx = (wrem % 5) * WS + t % WS;
  int tid = threadIdx.x;
  bf16_t* orow = out + (size_t)r * Cc;
  if (y >= 64 || xx >= 64) {
    for (int c = tid; c < Cc; c += 256) orow[c] = (bf16_t)0.0f;
    return;
  }
  const float* irow = x + (((size_t)b * 64 + y) * 64 + xx) * Cc;
  float v[3], s = 0.f, sq = 0.f;
#pragma unroll
  for (int i = 0; i < 3; ++i) {
    v[i] = irow[tid + 256 * i];
    s += v[i]; sq += v[i] * v[i];
  }
#pragma unroll
  for (int o = 16; o > 0; o >>= 1) {
    s  += __shfl_xor(s,  o, 32);
    sq += __shfl_xor(sq, o, 32);
  }
  __shared__ float red[16];
  int wave = tid >> 5, lane = tid & 31;
  if (lane == 0) { red[wave] = s; red[8 + wave] = sq; }
  __syncthreads();
  float ts = 0.f, tq = 0.f;
#pragma unroll
  for (int i = 0; i < 8; ++i) { ts += red[i]; tq += red[8 + i]; }
  float mu = ts * (1.0f / Cc);
  float rstd = rsqrtf(tq * (1.0f / Cc) - mu * mu + 1e-6f);
#pragma unroll
  for (int i = 0; i < 3; ++i) {
    int c = tid + 256 * i;
    orow[c] = (bf16_t)((v[i] - mu) * rstd * sc[c] + bi[c]);
  }
}

// =====================================================================
// LayerNorm2 (image-order tokens)
// =====================================================================
__global__ __launch_bounds__(256) void ln2_kernel(
    const float* __restrict__ x, const float* __restrict__ sc,
    const float* __restrict__ bi, bf16_t* __restrict__ out) {
  int r = blockIdx.x;
  int tid = threadIdx.x;
  const float* irow = x + (size_t)r * Cc;
  bf16_t* orow = out + (size_t)r * Cc;
  float v[3], s = 0.f, sq = 0.f;
#pragma unroll
  for (int i = 0; i < 3; ++i) {
    v[i] = irow[tid + 256 * i];
    s += v[i]; sq += v[i] * v[i];
  }
#pragma unroll
  for (int o = 16; o > 0; o >>= 1) {
    s  += __shfl_xor(s,  o, 32);
    sq += __shfl_xor(sq, o, 32);
  }
  __shared__ float red[16];
  int wave = tid >> 5, lane = tid & 31;
  if (lane == 0) { red[wave] = s; red[8 + wave] = sq; }
  __syncthreads();
  float ts = 0.f, tq = 0.f;
#pragma unroll
  for (int i = 0; i < 8; ++i) { ts += red[i]; tq += red[8 + i]; }
  float mu = ts * (1.0f / Cc);
  float rstd = rsqrtf(tq * (1.0f / Cc) - mu * mu + 1e-6f);
#pragma unroll
  for (int i = 0; i < 3; ++i) {
    int c = tid + 256 * i;
    orow[c] = (bf16_t)((v[i] - mu) * rstd * sc[c] + bi[c]);
  }
}

// =====================================================================
// bf16 WMMA GEMM: Out = A(MxK) @ Wt(NxK)^T + bias, fused epilogue.
// Block 256 thr (8 wave32), tile 128x128, wave tile 32x64, K-step 32,
// double-buffered LDS filled by the Tensor Data Mover (wave 0 issues the
// D#, TENSORcnt drained before the barrier); register-pipeline fallback.
// =====================================================================
constexpr int TILE_ELEMS = 128 * LDT;                 // one padded tile
constexpr unsigned SB_BYTE_OFF = 2 * TILE_ELEMS * 2;  // sB pool offset

template <int MODE>
__global__ __launch_bounds__(256) void gemm_bf16_kernel(
    const bf16_t* __restrict__ A, const bf16_t* __restrict__ Wt,
    const float* __restrict__ bias, int M, int N, int K,
    bf16_t* __restrict__ oQ, bf16_t* __restrict__ oK, bf16_t* __restrict__ oV,
    float* __restrict__ oF, const float* __restrict__ res,
    bf16_t* __restrict__ oB) {
  __shared__ bf16_t pool[4 * TILE_ELEMS];   // [2][128*LDT] A then [2][...] B
  // runtime-computed buffer pointers (no static-initialized LDS pointer array)
  auto sAp = [&](int buf) -> bf16_t* { return pool + buf * TILE_ELEMS; };
  auto sBp = [&](int buf) -> bf16_t* { return pool + (2 + buf) * TILE_ELEMS; };

  int tid  = threadIdx.x;
  int wave = tid >> 5, lane = tid & 31;
  int r0 = blockIdx.x * 128;
  int n0 = blockIdx.y * 128;
  int mw = wave >> 1, nw = wave & 1;          // 4x2 wave grid

  v8f acc[2][4] = {};
  int nk = K >> 5;

#ifdef USE_TDM
  unsigned int rowsA = (unsigned int)(M - r0);     // OOB rows read as zero
  auto issueTiles = [&](int kk0, int buf) {
    tdm_load_2d((unsigned)(buf * TILE_ELEMS * 2),
                A + (size_t)r0 * K + kk0, 32u, rowsA,
                (unsigned long long)K, 32u, 128u, 3u, 3u);
    tdm_load_2d(SB_BYTE_OFF + (unsigned)(buf * TILE_ELEMS * 2),
                Wt + (size_t)n0 * K + kk0, 32u, 128u,
                (unsigned long long)K, 32u, 128u, 3u, 3u);
  };
  if (wave == 0) {
    issueTiles(0, 0);
    __builtin_amdgcn_s_wait_tensorcnt(0);
  }
  __syncthreads();
  for (int t = 0; t < nk; ++t) {
    int buf = t & 1;
    bool more = (t + 1) < nk;
    if (more && wave == 0) issueTiles((t + 1) << 5, buf ^ 1);
    const bf16_t* sA = sAp(buf);
    const bf16_t* sB = sBp(buf);
    v16bf a0 = ldA(sA + (mw * 32) * LDT, LDT);
    v16bf a1 = ldA(sA + (mw * 32 + 16) * LDT, LDT);
#pragma unroll
    for (int j = 0; j < 4; ++j) {
      v16bf b = ldB_T(sB + (nw * 64 + j * 16) * LDT, LDT);
      acc[0][j] = wmma_bf16(a0, b, acc[0][j]);
      acc[1][j] = wmma_bf16(a1, b, acc[1][j]);
    }
    if (more && wave == 0) __builtin_amdgcn_s_wait_tensorcnt(0);
    __syncthreads();
  }
#else
  // register-pipelined fallback (global b128 -> reg -> LDS b128)
  int row0 = tid >> 2,         q0 = (tid & 3) * 8;
  int row1 = (tid + 256) >> 2, q1 = ((tid + 256) & 3) * 8;
  uint4 ra0, ra1, rb0, rb1;
  const uint4 z4 = {0u, 0u, 0u, 0u};
  auto loadRegs = [&](int kk0) {
    int g0 = r0 + row0, g1 = r0 + row1;
    ra0 = (g0 < M) ? *(const uint4*)(A + (size_t)g0 * K + kk0 + q0) : z4;
    ra1 = (g1 < M) ? *(const uint4*)(A + (size_t)g1 * K + kk0 + q1) : z4;
    rb0 = *(const uint4*)(Wt + (size_t)(n0 + row0) * K + kk0 + q0);
    rb1 = *(const uint4*)(Wt + (size_t)(n0 + row1) * K + kk0 + q1);
  };
  auto storeLds = [&](int buf) {
    *(uint4*)(sAp(buf) + row0 * LDT + q0) = ra0;
    *(uint4*)(sAp(buf) + row1 * LDT + q1) = ra1;
    *(uint4*)(sBp(buf) + row0 * LDT + q0) = rb0;
    *(uint4*)(sBp(buf) + row1 * LDT + q1) = rb1;
  };
  loadRegs(0);
  storeLds(0);
  __syncthreads();
  for (int t = 0; t < nk; ++t) {
    int buf = t & 1;
    bool more = (t + 1) < nk;
    if (more) loadRegs((t + 1) << 5);
    if (t + 2 < nk) {
      int pk = (t + 2) << 5;
      __builtin_prefetch(Wt + (size_t)(n0 + row0) * K + pk + q0, 0, 1);
      int g0 = r0 + row0;
      if (g0 < M) __builtin_prefetch(A + (size_t)g0 * K + pk + q0, 0, 1);
    }
    const bf16_t* sA = sAp(buf);
    const bf16_t* sB = sBp(buf);
    v16bf a0 = ldA(sA + (mw * 32) * LDT, LDT);
    v16bf a1 = ldA(sA + (mw * 32 + 16) * LDT, LDT);
#pragma unroll
    for (int j = 0; j < 4; ++j) {
      v16bf b = ldB_T(sB + (nw * 64 + j * 16) * LDT, LDT);
      acc[0][j] = wmma_bf16(a0, b, acc[0][j]);
      acc[1][j] = wmma_bf16(a1, b, acc[1][j]);
    }
    if (more) storeLds(buf ^ 1);
    __syncthreads();
  }
#endif

  // ---------------- epilogue ----------------
  int nl = lane & 15;
  int rbase = (lane >> 4) << 3;
#pragma unroll
  for (int ia = 0; ia < 2; ++ia) {
#pragma unroll
    for (int j = 0; j < 4; ++j) {
      int col = n0 + nw * 64 + j * 16 + nl;
      float bc = bias[col];
#pragma unroll
      for (int i = 0; i < 8; ++i) {
        int row = r0 + mw * 32 + ia * 16 + rbase + i;
        if (row >= M) continue;
        float v = acc[ia][j][i] + bc;
        if (MODE == 0) {
          int which = col / Cc, c2 = col - which * Cc;
          int head = c2 >> 6, d = c2 & 63;
          int wi = row / LWIN, t2 = row - wi * LWIN;
          bf16_t* dst = (which == 0) ? oQ : (which == 1) ? oK : oV;
          dst[(((size_t)wi * HEADS + head) * LWIN + t2) * HD + d] = (bf16_t)v;
        } else if (MODE == 1) {
          int wi = row / LWIN, t2 = row - wi * LWIN;
          int b = wi / 25, wrem = wi % 25;
          int y = (wrem / 5) * WS + t2 / WS;
          int xx = (wrem % 5) * WS + t2 % WS;
          if (y < 64 && xx < 64) {
            size_t idx = ((((size_t)b * 64 + y) * 64 + xx) * Cc) + col;
            oF[idx] = res[idx] + v;
          }
        } else if (MODE == 2) {
          float g = 0.5f * v * (1.0f + erff(v * 0.70710678118654752f));
          oB[(size_t)row * N + col] = (bf16_t)g;
        } else {
          size_t idx = (size_t)row * Cc + col;
          oF[idx] = res[idx] + v;
        }
      }
    }
  }
}

// =====================================================================
// Fused window attention, one block per (window, head).
// ~316 KB of the WGP's 320 KB LDS, conflict-free padded layouts.
// =====================================================================
__global__ __launch_bounds__(256) void attn_kernel(
    const bf16_t* __restrict__ Qb, const bf16_t* __restrict__ Kb,
    const bf16_t* __restrict__ Vb, const float* __restrict__ rel_h,
    const float* __restrict__ rel_w, bf16_t* __restrict__ Ytok) {
  extern __shared__ char smem[];
  bf16_t* sQ  = (bf16_t*)smem;                  // 224*72*2  = 32256
  bf16_t* sK  = (bf16_t*)(smem + 32256);        // 32256
  bf16_t* sVt = (bf16_t*)(smem + 64512);        // 64*232*2  = 29696
  float*  sS  = (float*)(smem + 94208);         // 224*228*4 = 204288
  float*  sBh = (float*)(smem + 298496);        // 224*14*4  = 12544
  float*  sBw = (float*)(smem + 311040);        // 12544 -> total 323584

  int wh = blockIdx.x;
  int wi = wh / HEADS, h = wh - wi * HEADS;
  int tid = threadIdx.x;
  int wave = tid >> 5, lane = tid & 31;
  size_t base = (size_t)wh * LWIN * HD;
  const uint4 z4 = {0u, 0u, 0u, 0u};

  // ---- stage Q,K (t-major, padded) and V (d-major) via b128 loads ----
  for (int e = tid; e < LP * 8; e += 256) {
    int t = e >> 3, d0 = (e & 7) * 8;
    uint4 qv = z4, kv = z4, vv = z4;
    if (t < LWIN) {
      size_t o = base + (size_t)t * HD + d0;
      qv = *(const uint4*)(Qb + o);
      kv = *(const uint4*)(Kb + o);
      vv = *(const uint4*)(Vb + o);
    }
    *(uint4*)(sQ + t * LDQ + d0) = qv;
    *(uint4*)(sK + t * LDQ + d0) = kv;
    bf16_t tmp[8];
    *(uint4*)tmp = vv;
#pragma unroll
    for (int j = 0; j < 8; ++j) sVt[(d0 + j) * LDV + t] = tmp[j];
  }
  __syncthreads();

  // ---- rel-pos bias tables, padded to 224 rows (zeros) ----
  for (int e = tid; e < LP * WS * 2; e += 256) {
    int which = e / (LP * WS);
    int rem = e - which * (LP * WS);
    int t = rem / WS, kk = rem - t * WS;
    float s = 0.f;
    if (t < LWIN) {
      int qc = which ? (t % WS) : (t / WS);
      const float* rp = (which ? rel_w : rel_h) + (qc - kk + WS - 1) * HD;
      const bf16_t* qv = sQ + t * LDQ;
#pragma unroll
      for (int d = 0; d < HD; ++d) s += (float)qv[d] * rp[d];
    }
    (which ? sBw : sBh)[t * WS + kk] = s;
  }
  __syncthreads();

  // ---- S = scale * Q K^T + bias (masked); 14x14 tiles over 8 waves ----
  int nl = lane & 15;
  int rbase = (lane >> 4) << 3;
  for (int tt = wave; tt < 196; tt += 8) {
    int mi = tt / 14, ni = tt - mi * 14;
    v8f acc = {};
#pragma unroll
    for (int kk = 0; kk < 2; ++kk) {
      v16bf a = ldA(sQ + (mi * 16) * LDQ + kk * 32, LDQ);
      v16bf b = ldB_T(sK + (ni * 16) * LDQ + kk * 32, LDQ);
      acc = wmma_bf16(a, b, acc);
    }
    int coln = ni * 16 + nl;
    int rtop = mi * 16 + rbase;
    float* sp = sS + rtop * LDS_S + coln;
    if (coln >= LWIN) {
#pragma unroll
      for (int i = 0; i < 8; ++i) sp[i * LDS_S] = -1.0e30f;
    } else {
      int kh = coln / WS;
      int kw = coln - kh * WS;
      const float* bh = sBh + rtop * WS + kh;
      const float* bw = sBw + rtop * WS + kw;
#pragma unroll
      for (int i = 0; i < 8; ++i)
        sp[i * LDS_S] = acc[i] * 0.125f + bh[i * WS] + bw[i * WS];
    }
  }
  __syncthreads();

  // ---- softmax (wave per row, 224 = 7*32 cols) ----
  for (int r = wave; r < LP; r += 8) {
    float* row = sS + r * LDS_S;
    float vals[7], m = -1.0e30f;
#pragma unroll
    for (int i = 0; i < 7; ++i) { vals[i] = row[lane + 32 * i]; m = fmaxf(m, vals[i]); }
#pragma unroll
    for (int o = 16; o > 0; o >>= 1) m = fmaxf(m, __shfl_xor(m, o, 32));
    float s = 0.f;
#pragma unroll
    for (int i = 0; i < 7; ++i) { vals[i] = __expf(vals[i] - m); s += vals[i]; }
#pragma unroll
    for (int o = 16; o > 0; o >>= 1) s += __shfl_xor(s, o, 32);
    float inv = 1.0f / s;
#pragma unroll
    for (int i = 0; i < 7; ++i) row[lane + 32 * i] = vals[i] * inv;
  }
  __syncthreads();

  // ---- O = P V -> token-major Y ----
  for (int tt = wave; tt < 56; tt += 8) {
    int mi = tt >> 2, nj = tt & 3;
    v8f acc = {};
#pragma unroll
    for (int kk = 0; kk < 7; ++kk) {
      v16bf a = ldA_f32(sS + (mi * 16) * LDS_S + kk * 32, LDS_S);
      v16bf b = ldB_T(sVt + (nj * 16) * LDV + kk * 32, LDV);
      acc = wmma_bf16(a, b, acc);
    }
#pragma unroll
    for (int i = 0; i < 8; ++i) {
      int m = mi * 16 + rbase + i;
      if (m < LWIN)
        Ytok[((size_t)wi * LWIN + m) * Cc + h * HD + nj * 16 + nl] = (bf16_t)acc[i];
    }
  }
}

// =====================================================================
// host-side launcher
// =====================================================================
extern "C" void kernel_launch(void* const* d_in, const int* in_sizes, int n_in,
                              void* d_out, int out_size, void* d_ws, size_t ws_size,
                              hipStream_t stream) {
  const float* x      = (const float*)d_in[0];
  const float* n1s    = (const float*)d_in[1];
  const float* n1b    = (const float*)d_in[2];
  const float* qkv_w  = (const float*)d_in[3];
  const float* qkv_b  = (const float*)d_in[4];
  const float* rel_h  = (const float*)d_in[5];
  const float* rel_w  = (const float*)d_in[6];
  const float* proj_w = (const float*)d_in[7];
  const float* proj_b = (const float*)d_in[8];
  const float* n2s    = (const float*)d_in[9];
  const float* n2b    = (const float*)d_in[10];
  const float* fc1_w  = (const float*)d_in[11];
  const float* fc1_b  = (const float*)d_in[12];
  const float* fc2_w  = (const float*)d_in[13];
  const float* fc2_b  = (const float*)d_in[14];
  float* out = (float*)d_out;

  char* ws = (char*)d_ws;
  auto up = [](size_t v) { return (v + 255) & ~(size_t)255; };
  size_t off = 0;
  bf16_t* wqkv = (bf16_t*)(ws + off); off += up((size_t)Cc * C3 * 2);
  bf16_t* wprj = (bf16_t*)(ws + off); off += up((size_t)Cc * Cc * 2);
  bf16_t* wf1  = (bf16_t*)(ws + off); off += up((size_t)Cc * HID * 2);
  bf16_t* wf2  = (bf16_t*)(ws + off); off += up((size_t)HID * Cc * 2);
  bf16_t* Xtok = (bf16_t*)(ws + off); off += up((size_t)MQKV * Cc * 2);
  size_t big = off;
  bf16_t* Qb   = (bf16_t*)(ws + off); off += up((size_t)NWH * LWIN * HD * 2);
  bf16_t* Kb   = (bf16_t*)(ws + off); off += up((size_t)NWH * LWIN * HD * 2);
  bf16_t* Vb   = (bf16_t*)(ws + off); off += up((size_t)NWH * LWIN * HD * 2);
  bf16_t* Ytok = (bf16_t*)(ws + off); off += up((size_t)MQKV * Cc * 2);
  float*  xres = (float*)(ws + off);  off += up((size_t)MTOK * Cc * 4);
  bf16_t* h2   = Xtok;                           // reuse
  bf16_t* m1   = (bf16_t*)(ws + big);            // reuse Q..Y region
  (void)ws_size; (void)in_sizes; (void)n_in; (void)out_size;

  transpose_cvt_kernel<<<dim3(C3 / 32, Cc / 32), 256, 0, stream>>>(qkv_w, wqkv, Cc, C3);
  transpose_cvt_kernel<<<dim3(Cc / 32, Cc / 32), 256, 0, stream>>>(proj_w, wprj, Cc, Cc);
  transpose_cvt_kernel<<<dim3(HID / 32, Cc / 32), 256, 0, stream>>>(fc1_w, wf1, Cc, HID);
  transpose_cvt_kernel<<<dim3(Cc / 32, HID / 32), 256, 0, stream>>>(fc2_w, wf2, HID, Cc);
  ln1_window_kernel<<<MQKV, 256, 0, stream>>>(x, n1s, n1b, Xtok);
  gemm_bf16_kernel<0><<<dim3((MQKV + 127) / 128, C3 / 128), 256, 0, stream>>>(
      Xtok, wqkv, qkv_b, MQKV, C3, Cc, Qb, Kb, Vb, nullptr, nullptr, nullptr);
  attn_kernel<<<NWH, 256, 323584, stream>>>(Qb, Kb, Vb, rel_h, rel_w, Ytok);
  gemm_bf16_kernel<1><<<dim3((MQKV + 127) / 128, Cc / 128), 256, 0, stream>>>(
      Ytok, wprj, proj_b, MQKV, Cc, Cc, nullptr, nullptr, nullptr, xres, x, nullptr);
  ln2_kernel<<<MTOK, 256, 0, stream>>>(xres, n2s, n2b, h2);
  gemm_bf16_kernel<2><<<dim3(MTOK / 128, HID / 128), 256, 0, stream>>>(
      h2, wf1, fc1_b, MTOK, HID, Cc, nullptr, nullptr, nullptr, nullptr, nullptr, m1);
  gemm_bf16_kernel<3><<<dim3(MTOK / 128, Cc / 128), 256, 0, stream>>>(
      m1, wf2, fc2_b, MTOK, Cc, HID, nullptr, nullptr, nullptr, out, xres, nullptr);
}
